// DeformAtten1D_59270548685455
// MI455X (gfx1250) — matrix operand order repaired and checked by reading, not compiled, not run on hardware
//
#include <hip/hip_runtime.h>
#include <hip/hip_bf16.h>
#include <cstddef>
#include <cstdint>

// ---------------------------------------------------------------------------
// DeformAtten1D for MI455X (gfx1250, wave32, WMMA bf16 16x16x32)
//
// Shapes: B=16, L=2048, C=1024, H=16, G=4, K=5, GC=256, DH=64
// Offset network collapsed: off = gq (*) Weff, Weff[c,k] = sum_o w2[o] W1[o,c,k]
// (removes the 85.9-GFLOP grouped conv; remaining ~283 GFLOP is bf16 WMMA GEMM).
//
// LDS tiles are FRAGMENT-MAJOR (ISA 7.12.2 lane maps inverted at staging time)
// so each lane reads its whole v16bf WMMA operand as one contiguous 32B LDS
// access (2x ds_load_b128). Because 8 consecutive k from an 8-aligned base map
// to 8 consecutive fragment elements of a single lane, bf16 tiles are staged
// with CDNA5 GLOBAL_LOAD_ASYNC_TO_LDS_B128 (ASYNCcnt; VGPR-bypassing copies),
// waited with s_wait_asynccnt before the workgroup barrier.
// ---------------------------------------------------------------------------

typedef __bf16 v16bf __attribute__((ext_vector_type(16)));
typedef float  v8f   __attribute__((ext_vector_type(8)));
typedef __bf16 bf16x4 __attribute__((ext_vector_type(4)));

// exact pointee type expected by the async-load builtin (from hipcc diagnostic)
typedef int v4i_vs __attribute__((vector_size(16)));
typedef __attribute__((address_space(1))) v4i_vs as1_v4i;
typedef __attribute__((address_space(3))) v4i_vs as3_v4i;

constexpr int B_   = 16;
constexpr int L_   = 2048;
constexpr int C_   = 1024;
constexpr int H_   = 16;
constexpr int G_   = 4;
constexpr int K_   = 5;
constexpr int GC_  = 256;   // C/G
constexpr int DH_  = 64;    // C/H
constexpr int PAD_ = 2;
constexpr int L4_  = L_ + 2 * PAD_;   // 2052
constexpr int M_   = B_ * L_;         // 32768 token-major rows

// A-operand lane map (16x32 bf16): lane = (m&15) + 16*halfA(k), elem = eA(k)
__device__ __forceinline__ int halfA_of_k(int k) { return (k >> 3) & 1; }
__device__ __forceinline__ int eA_of_k(int k) { return (k & 7) + ((k & 16) ? 8 : 0); }
// B-operand lane map (32x16 bf16): lane = (n&15) + 16*halfA(k), elem = eA(k)

// ---- CDNA5 async global->LDS copy (16B per lane) + ASYNCcnt wait ----------
__device__ __forceinline__ void async_load_b128(const void* g, void* l) {
#if __has_builtin(__builtin_amdgcn_global_load_async_to_lds_b128)
  __builtin_amdgcn_global_load_async_to_lds_b128(
      (as1_v4i*)(uintptr_t)g,
      (as3_v4i*)(uint32_t)(uintptr_t)l, 0, 0);
#else
  unsigned loff = (unsigned)(uintptr_t)l;   // LDS_ADDR = flat[31:0]
  asm volatile("global_load_async_to_lds_b128 %0, %1, off"
               :: "v"(loff), "v"(g) : "memory");
#endif
}
__device__ __forceinline__ void wait_async() {
#if __has_builtin(__builtin_amdgcn_s_wait_asynccnt)
  __builtin_amdgcn_s_wait_asynccnt(0);
#else
  asm volatile("s_wait_asynccnt 0" ::: "memory");
#endif
}

// ---------------------------------------------------------------------------
__global__ void cvt_bf16_kernel(const float* __restrict__ src,
                                __bf16* __restrict__ dst, int n) {
  int t = blockIdx.x * blockDim.x + threadIdx.x;
  if (t < n) dst[t] = (__bf16)src[t];
}

// Fold the two offset convs: Weff[c*K+k] = sum_o w2[o]*W1[o,c,k];
// beff (at Weff[GC*K]) = sum_o w2[o]*b1[o] + b2
__global__ void weff_kernel(const float* __restrict__ W1, const float* __restrict__ b1,
                            const float* __restrict__ W2, const float* __restrict__ b2,
                            float* __restrict__ weff) {
  int t = blockIdx.x * blockDim.x + threadIdx.x;
  if (t < GC_ * K_) {
    int c = t / K_, k = t % K_;
    float s = 0.f;
    for (int o = 0; o < GC_; ++o)
      s += W2[o] * W1[((size_t)o * GC_ + c) * K_ + k];
    weff[t] = s;
  } else if (t == GC_ * K_) {
    float s = b2[0];
    for (int o = 0; o < GC_; ++o) s += W2[o] * b1[o];
    weff[t] = s;
  }
}

// ---------------------------------------------------------------------------
// Tiled WMMA GEMM:  C[M,N] = A[M,K] * Bw[N,K]^T + bias (+ rpb)
// 128 threads = 4 wave32; 64(M) x 128(N) tile, K-chunks of 32.
// ---------------------------------------------------------------------------
template <typename AT, typename OT, bool ADD_RPB>
__global__ __launch_bounds__(128) void gemm_kernel(
    const AT* __restrict__ A, const __bf16* __restrict__ Bw,
    const float* __restrict__ bias, const float* __restrict__ rpb,
    OT* __restrict__ Cout, int Mdim, int Kdim, int Ndim) {
  __shared__ __align__(32) __bf16 AF[4][32][16];   // 4 KB fragment-major
  __shared__ __align__(32) __bf16 BF[8][32][16];   // 8 KB fragment-major

  const int tid  = threadIdx.x;
  const int wave = tid >> 5;
  const int lane = tid & 31;
  const int half = lane >> 4;
  const int ml   = lane & 15;
  const int n0   = blockIdx.x * 128;
  const int m0   = blockIdx.y * 64;

  v8f acc[8] = {};

  for (int k0 = 0; k0 < Kdim; k0 += 32) {
    // ---- stage A tile (64m x 32k)
    if constexpr (__is_same(AT, __bf16)) {
      // async copy: 8 consecutive k == 8 consecutive fragment elems of 1 lane
#pragma unroll
      for (int it = 0; it < 2; ++it) {
        int idx = tid + it * 128;         // 256 chunks of 16B
        int r   = idx >> 2;               // m in [0,64)
        int c8  = (idx & 3) << 3;         // k in {0,8,16,24}
        async_load_b128(A + (size_t)(m0 + r) * Kdim + k0 + c8,
                        &AF[r >> 4][(r & 15) + (((c8 >> 3) & 1) << 4)]
                           [(c8 & 16) ? 8 : 0]);
      }
    } else {
      // fp32 source: load+convert via VGPRs
#pragma unroll
      for (int it = 0; it < 4; ++it) {
        int idx = tid + it * 128;
        int r   = idx >> 3;               // m in [0,64)
        int c   = (idx & 7) << 2;         // k in {0,4,...,28}
        const AT* s = A + (size_t)(m0 + r) * Kdim + k0 + c;
        bf16x4 t;
        t.x = (__bf16)(float)s[0];
        t.y = (__bf16)(float)s[1];
        t.z = (__bf16)(float)s[2];
        t.w = (__bf16)(float)s[3];
        *(bf16x4*)&AF[r >> 4][(r & 15) + (halfA_of_k(c) << 4)][eA_of_k(c)] = t;
      }
      if (k0 + 32 < Kdim)  // CDNA5 global_prefetch_b8 of the next K-chunk
        __builtin_prefetch(A + (size_t)(m0 + (tid >> 1)) * Kdim + k0 + 32 + (tid & 1) * 16, 0, 1);
    }
    // ---- stage B tile (128n x 32k) via async copies
#pragma unroll
    for (int it = 0; it < 4; ++it) {
      int idx = tid + it * 128;           // 512 chunks of 16B
      int r   = idx >> 2;                 // n in [0,128)
      int c8  = (idx & 3) << 3;           // k in {0,8,16,24}
      async_load_b128(Bw + (size_t)(n0 + r) * Kdim + k0 + c8,
                      &BF[r >> 4][(r & 15) + (((c8 >> 3) & 1) << 4)][(c8 & 16) ? 8 : 0]);
    }
    wait_async();
    __syncthreads();

    v16bf af = *(const v16bf*)&AF[wave][lane][0];   // 2x ds_load_b128
#pragma unroll
    for (int nt = 0; nt < 8; ++nt) {
      v16bf bfr = *(const v16bf*)&BF[nt][lane][0];
      acc[nt] = __builtin_amdgcn_wmma_f32_16x16x32_bf16(
          false, af, false, bfr, (short)0, acc[nt], false, false);
    }
    __syncthreads();
  }

  // epilogue: D layout -> m = r + 8*half (+16*wave), n = ml (+16*nt)
#pragma unroll
  for (int nt = 0; nt < 8; ++nt) {
#pragma unroll
    for (int r = 0; r < 8; ++r) {
      int row = m0 + wave * 16 + half * 8 + r;
      int col = n0 + nt * 16 + ml;
      float v = acc[nt][r] + bias[col];
      if (ADD_RPB) v += rpb[(size_t)col * L_ + (row & (L_ - 1))];
      Cout[(size_t)row * Ndim + col] = (OT)v;
    }
  }
}

// ---------------------------------------------------------------------------
// Offset + tanh + sampling-position kernel (collapsed offset network)
// ---------------------------------------------------------------------------
__global__ __launch_bounds__(256) void offset_kernel(
    const float* __restrict__ qbl, const float* __restrict__ weff,
    float* __restrict__ pos) {
  __shared__ float w[GC_ * K_ + 1];
  for (int i = threadIdx.x; i <= GC_ * K_; i += 256) w[i] = weff[i];
  __syncthreads();

  int t  = blockIdx.x * 256 + threadIdx.x;   // t = bg * L + j
  int bg = t >> 11;
  int j  = t & (L_ - 1);
  int b  = bg >> 2;
  int g  = bg & 3;

  float off = 0.f;
  if (j >= 2) {
    int l = j - 2;
    float acc = w[GC_ * K_];  // beff
    for (int k = 0; k < K_; ++k) {
      int lp = l + k - PAD_;
      if (lp < 0 || lp >= L_) continue;
      const float* qrow = qbl + ((size_t)(b * L_ + lp)) * C_ + g * GC_;
      for (int c = 0; c < GC_; ++c) acc += qrow[c] * w[c * K_ + k];
    }
    off = tanhf(acc) * (float)K_;
  }
  pos[t] = ((float)j + off) * ((float)L_ / (float)(L4_ - 1)) - 0.5f;
}

// ---------------------------------------------------------------------------
// Bilinear deformable sampling -> bf16 token-major features
// ---------------------------------------------------------------------------
__global__ __launch_bounds__(256) void sample_kernel(
    const float* __restrict__ x, const float* __restrict__ pos,
    __bf16* __restrict__ xs) {
  int t   = blockIdx.x * 256 + threadIdx.x;  // one thread per 4 channels
  int row = t >> 8;
  int c   = (t & 255) << 2;
  int b   = row >> 11;
  int j   = row & (L_ - 1);
  int g   = c >> 8;

  float p  = pos[(b * G_ + g) * L_ + j];
  float fi = floorf(p);
  int   i0 = (int)fi;
  float w1 = p - fi;
  float w0 = 1.f - w1;

  float4 a = make_float4(0.f, 0.f, 0.f, 0.f);
  float4 bb = make_float4(0.f, 0.f, 0.f, 0.f);
  if (i0 >= 0 && i0 < L_)
    a = *(const float4*)(x + ((size_t)b * L_ + i0) * C_ + c);
  int i1 = i0 + 1;
  if (i1 >= 0 && i1 < L_)
    bb = *(const float4*)(x + ((size_t)b * L_ + i1) * C_ + c);

  bf16x4 r;
  r.x = (__bf16)(a.x * w0 + bb.x * w1);
  r.y = (__bf16)(a.y * w0 + bb.y * w1);
  r.z = (__bf16)(a.z * w0 + bb.z * w1);
  r.w = (__bf16)(a.w * w0 + bb.w * w1);
  *(bf16x4*)(xs + (size_t)row * C_ + c) = r;
}

// ---------------------------------------------------------------------------
// Per-(b,h) attention, fragment-major LDS throughout.
//   scores = q^T k over L (WMMA), fp32 softmax, out = attn * v (WMMA)
// ---------------------------------------------------------------------------
struct P1 { __bf16 QF[4][32][16]; __bf16 KF[4][32][16]; };  // 8 KB
union AttnU {
  P1 p1;
  float sc[64][65];            // 16640 B (aliases dead QF/KF)
  __bf16 VF[4][2][32][16];     // 8 KB  (aliases dead scores)
};

__global__ __launch_bounds__(128) void attn_kernel(
    const float* __restrict__ qbl, const __bf16* __restrict__ kbl,
    const __bf16* __restrict__ vbl, __bf16* __restrict__ att) {
  __shared__ __align__(32) AttnU u;
  __shared__ __align__(32) __bf16 AF2[4][2][32][16];  // attn probs, A-frag layout

  const int tid  = threadIdx.x;
  const int wave = tid >> 5;
  const int lane = tid & 31;
  const int half = lane >> 4;
  const int ml   = lane & 15;
  const int bh   = blockIdx.x;
  const int b    = bh >> 4;
  const int h    = bh & 15;
  const size_t rowbase = (size_t)b * L_;
  const int hc   = h * DH_;

  // ---- phase 1: scores[i][j] = sum_l q[i,l] k[j,l], K-chunks of 32 over L
  v8f acc[4] = {};
  for (int l0 = 0; l0 < L_; l0 += 32) {
    // transpose staging: consecutive global elems hit different lanes'
    // fragments, so this path stays scalar (async b128 not applicable)
#pragma unroll
    for (int it = 0; it < 4; ++it) {
      int idx = tid + it * 128;
      int r = idx >> 4;          // lrel in [0,32)  (WMMA K index)
      int c = (idx & 15) << 2;   // i/j base in [0,64)
      const float* qs = qbl + (rowbase + l0 + r) * C_ + hc + c;
      int eA = eA_of_k(r), hA = halfA_of_k(r) << 4, it4 = c >> 4;
      u.p1.QF[it4][(c & 15) + 0 + hA][eA] = (__bf16)qs[0];
      u.p1.QF[it4][(c & 15) + 1 + hA][eA] = (__bf16)qs[1];
      u.p1.QF[it4][(c & 15) + 2 + hA][eA] = (__bf16)qs[2];
      u.p1.QF[it4][(c & 15) + 3 + hA][eA] = (__bf16)qs[3];
      const __bf16* ks = kbl + (rowbase + l0 + r) * C_ + hc + c;
      int eB = eA_of_k(r), hB = halfA_of_k(r) << 4;
      u.p1.KF[it4][(c & 15) + 0 + hB][eB] = ks[0];
      u.p1.KF[it4][(c & 15) + 1 + hB][eB] = ks[1];
      u.p1.KF[it4][(c & 15) + 2 + hB][eB] = ks[2];
      u.p1.KF[it4][(c & 15) + 3 + hB][eB] = ks[3];
    }
    __syncthreads();

    v16bf af = *(const v16bf*)&u.p1.QF[wave][lane][0];
#pragma unroll
    for (int nt = 0; nt < 4; ++nt) {
      v16bf bfr = *(const v16bf*)&u.p1.KF[nt][lane][0];
      acc[nt] = __builtin_amdgcn_wmma_f32_16x16x32_bf16(
          false, af, false, bfr, (short)0, acc[nt], false, false);
    }
    __syncthreads();
  }

  // ---- scale + fp32 scores into LDS
#pragma unroll
  for (int nt = 0; nt < 4; ++nt)
#pragma unroll
    for (int r = 0; r < 8; ++r)
      u.sc[wave * 16 + half * 8 + r][nt * 16 + ml] = acc[nt][r] * 0.03125f; // C^-0.5
  __syncthreads();

  // ---- softmax over j; emit probabilities directly in A-fragment layout
  if (tid < 64) {
    int i = tid;
    float mx = -1e30f;
    for (int j = 0; j < 64; ++j) mx = fmaxf(mx, u.sc[i][j]);
    float s = 0.f;
    for (int j = 0; j < 64; ++j) {
      float e = __expf(u.sc[i][j] - mx);
      u.sc[i][j] = e;
      s += e;
    }
    float inv = 1.f / s;
    for (int j = 0; j < 64; ++j) {
      int kc = j >> 5, jr = j & 31;
      AF2[i >> 4][kc][(i & 15) + (halfA_of_k(jr) << 4)][eA_of_k(jr)] =
          (__bf16)(u.sc[i][j] * inv);
    }
  }
  __syncthreads();

  // ---- phase 3: out[i,l] = sum_j attn[i,j] v[j,l], N-chunks of 64 over L
  for (int l0 = 0; l0 < L_; l0 += 64) {
    // V tile: 8 consecutive j -> 8 consecutive frag elems => async b128
#pragma unroll
    for (int it = 0; it < 4; ++it) {
      int idx = tid + it * 128;     // 512 chunks of 16B
      int r   = idx >> 3;           // lrel in [0,64)  (WMMA N index)
      int c8  = (idx & 7) << 3;     // j in {0,8,...,56} (WMMA K index)
      int jr  = c8 & 31;
      async_load_b128(vbl + (rowbase + l0 + r) * C_ + hc + c8,
                      &u.VF[r >> 4][c8 >> 5][(r & 15) + (((jr >> 3) & 1) << 4)]
                           [(jr & 16) ? 8 : 0]);
    }
    wait_async();
    __syncthreads();

    v8f o[4] = {};
#pragma unroll
    for (int kc = 0; kc < 2; ++kc) {
      v16bf af2 = *(const v16bf*)&AF2[wave][kc][lane][0];
#pragma unroll
      for (int nt = 0; nt < 4; ++nt) {
        v16bf bfr = *(const v16bf*)&u.VF[nt][kc][lane][0];
        o[nt] = __builtin_amdgcn_wmma_f32_16x16x32_bf16(
            false, af2, false, bfr, (short)0, o[nt], false, false);
      }
    }
    // scatter: att[b, l, h*DH + i]
#pragma unroll
    for (int nt = 0; nt < 4; ++nt)
#pragma unroll
      for (int r = 0; r < 8; ++r)
        att[(rowbase + l0 + nt * 16 + ml) * C_ + hc + wave * 16 + half * 8 + r] =
            (__bf16)o[nt][r];
    __syncthreads();
  }
}

// ---------------------------------------------------------------------------
extern "C" void kernel_launch(void* const* d_in, const int* in_sizes, int n_in,
                              void* d_out, int out_size, void* d_ws, size_t ws_size,
                              hipStream_t stream) {
  (void)in_sizes; (void)n_in; (void)out_size; (void)ws_size;

  const float* x     = (const float*)d_in[0];
  const float* Wq    = (const float*)d_in[1];
  const float* bq    = (const float*)d_in[2];
  const float* Wk    = (const float*)d_in[3];
  const float* bk    = (const float*)d_in[4];
  const float* Wv    = (const float*)d_in[5];
  const float* bv    = (const float*)d_in[6];
  const float* Woff1 = (const float*)d_in[7];
  const float* boff1 = (const float*)d_in[8];
  const float* Woff2 = (const float*)d_in[9];
  const float* boff2 = (const float*)d_in[10];
  const float* rpb   = (const float*)d_in[11];
  const float* Wout  = (const float*)d_in[12];
  const float* bout  = (const float*)d_in[13];
  float* out = (float*)d_out;

  char* ws = (char*)d_ws;
  size_t off = 0;
  auto alloc = [&](size_t bytes) -> void* {
    void* p = ws + off;
    off = (off + bytes + 255) & ~(size_t)255;
    return p;
  };
  float*  qbl  = (float*)alloc((size_t)M_ * C_ * sizeof(float));
  __bf16* wqb  = (__bf16*)alloc((size_t)C_ * C_ * sizeof(__bf16));
  __bf16* wkb  = (__bf16*)alloc((size_t)C_ * C_ * sizeof(__bf16));
  __bf16* wvb  = (__bf16*)alloc((size_t)C_ * C_ * sizeof(__bf16));
  __bf16* wob  = (__bf16*)alloc((size_t)C_ * C_ * sizeof(__bf16));
  float*  weff = (float*)alloc((GC_ * K_ + 1) * sizeof(float));
  float*  pos  = (float*)alloc((size_t)B_ * G_ * L_ * sizeof(float));
  __bf16* xs   = (__bf16*)alloc((size_t)M_ * C_ * sizeof(__bf16));
  __bf16* kbl  = (__bf16*)alloc((size_t)M_ * C_ * sizeof(__bf16));
  __bf16* vbl  = (__bf16*)alloc((size_t)M_ * C_ * sizeof(__bf16));
  __bf16* att  = (__bf16*)alloc((size_t)M_ * C_ * sizeof(__bf16));

  const int nw = C_ * C_;
  cvt_bf16_kernel<<<nw / 256, 256, 0, stream>>>(Wq, wqb, nw);
  cvt_bf16_kernel<<<nw / 256, 256, 0, stream>>>(Wk, wkb, nw);
  cvt_bf16_kernel<<<nw / 256, 256, 0, stream>>>(Wv, wvb, nw);
  cvt_bf16_kernel<<<nw / 256, 256, 0, stream>>>(Wout, wob, nw);
  weff_kernel<<<(GC_ * K_ + 1 + 255) / 256, 256, 0, stream>>>(Woff1, boff1, Woff2, boff2, weff);

  dim3 gg(C_ / 128, M_ / 64);
  // q = x * Wq^T + bq  (fp32 out: feeds tanh-offset network + attention)
  gemm_kernel<float, float, false><<<gg, 128, 0, stream>>>(x, wqb, bq, nullptr, qbl, M_, C_, C_);
  // collapsed offset network -> sampling positions
  offset_kernel<<<(B_ * G_ * L_) / 256, 256, 0, stream>>>(qbl, weff, pos);
  // deformable bilinear sampling
  sample_kernel<<<((size_t)M_ * C_ / 4) / 256, 256, 0, stream>>>(x, pos, xs);
  // k = xs * Wk^T + bk ; v = xs * Wv^T + bv + rpb
  gemm_kernel<__bf16, __bf16, false><<<gg, 128, 0, stream>>>(xs, wkb, bk, nullptr, kbl, M_, C_, C_);
  gemm_kernel<__bf16, __bf16, true><<<gg, 128, 0, stream>>>(xs, wvb, bv, rpb, vbl, M_, C_, C_);
  // attention (scores WMMA + softmax + attn*V WMMA)
  attn_kernel<<<B_ * H_, 128, 0, stream>>>(qbl, kbl, vbl, att);
  // final projection
  gemm_kernel<__bf16, float, false><<<gg, 128, 0, stream>>>(att, wob, bout, nullptr, out, M_, C_, C_);
}
